// SNN_48739288875143
// MI455X (gfx1250) — matrix operand order repaired and checked by reading, not compiled, not run on hardware
//
#include <hip/hip_runtime.h>
#include <hip/hip_bf16.h>

// ---------------------------------------------------------------------------
// SNN (SLAYER-style): conv(s4,p1,k5) x3 + CUBA LIF, then dense 256->2056->11.
// MI455X plan: HBM-bound (~250MB traffic); direct convs for the sparse-event
// front end, v_wmma_f32_16x16x32_f16 for the two dense GEMMs (spikes are 0/1
// so f16 A/B + f32 accum is lossless-enough), CUBA scans fused with layout
// conversion to K-contiguous f16 transposed activations for WMMA B-fragments.
// ---------------------------------------------------------------------------

typedef __attribute__((ext_vector_type(16))) _Float16 v16h;
typedef __attribute__((ext_vector_type(8)))  _Float16 v8h;
typedef __attribute__((ext_vector_type(8)))  float    v8f;

#define SNN_T 64

// ---------------- weight norm: w = g * v / ||v||  (per output row) ----------
__global__ void snn_wnorm_f32(const float* __restrict__ v, const float* __restrict__ g,
                              float* __restrict__ out, int L) {
  __shared__ float red[256];
  const int o = blockIdx.x;
  const float* vr = v + (size_t)o * L;
  float s = 0.f;
  for (int i = threadIdx.x; i < L; i += blockDim.x) { float x = vr[i]; s += x * x; }
  red[threadIdx.x] = s;
  __syncthreads();
  for (int st = blockDim.x >> 1; st > 0; st >>= 1) {
    if ((int)threadIdx.x < st) red[threadIdx.x] += red[threadIdx.x + st];
    __syncthreads();
  }
  const float scale = g[o] / sqrtf(red[0]);
  for (int i = threadIdx.x; i < L; i += blockDim.x) out[(size_t)o * L + i] = vr[i] * scale;
}

// f16 output, padded to (Op rows x Lp cols); rows >= O and cols >= L are zero.
__global__ void snn_wnorm_f16(const float* __restrict__ v, const float* __restrict__ g,
                              _Float16* __restrict__ out, int O, int L, int Lp) {
  __shared__ float red[256];
  const int o = blockIdx.x;
  _Float16* orow = out + (size_t)o * Lp;
  if (o >= O) {  // uniform per block: zero the padding row
    for (int i = threadIdx.x; i < Lp; i += blockDim.x) orow[i] = (_Float16)0.f;
    return;
  }
  const float* vr = v + (size_t)o * L;
  float s = 0.f;
  for (int i = threadIdx.x; i < L; i += blockDim.x) { float x = vr[i]; s += x * x; }
  red[threadIdx.x] = s;
  __syncthreads();
  for (int st = blockDim.x >> 1; st > 0; st >>= 1) {
    if ((int)threadIdx.x < st) red[threadIdx.x] += red[threadIdx.x + st];
    __syncthreads();
  }
  const float scale = g[o] / sqrtf(red[0]);
  for (int i = threadIdx.x; i < Lp; i += blockDim.x)
    orow[i] = (i < L) ? (_Float16)(vr[i] * scale) : (_Float16)0.f;
}

// ---------------- direct conv, k=5 stride=4 pad=1, per-timestep ------------
// x: (B, Cin, Hin, Win, T)  w: (Cout, Cin, 5, 5)  z: (B, Cout, Hout, Wout, T)
// grid: (Hout*Wout, B*Cout), block: 64 threads over T (coalesced over t).
__global__ void snn_conv(const float* __restrict__ x, const float* __restrict__ w,
                         float* __restrict__ z, int Cin, int Hin, int Win,
                         int Cout, int Hout, int Wout) {
  const int t   = threadIdx.x;           // 0..63
  const int pix = blockIdx.x;
  const int wo  = pix % Wout, ho = pix / Wout;
  const int bo  = blockIdx.y;            // b*Cout + o
  const int o   = bo % Cout, b = bo / Cout;
  float acc = 0.f;
  const float* wf = w + (size_t)o * Cin * 25;
  for (int ci = 0; ci < Cin; ++ci) {
    const float* xp = x + ((size_t)(b * Cin + ci) * Hin) * Win * SNN_T;
    const float* wc = wf + ci * 25;
#pragma unroll
    for (int kh = 0; kh < 5; ++kh) {
      const int h = ho * 4 - 1 + kh;
      if ((unsigned)h >= (unsigned)Hin) continue;
#pragma unroll
      for (int kw = 0; kw < 5; ++kw) {
        const int wp = wo * 4 - 1 + kw;
        if ((unsigned)wp >= (unsigned)Win) continue;
        acc += xp[((size_t)h * Win + wp) * SNN_T + t] * wc[kh * 5 + kw];
      }
    }
  }
  z[(((size_t)bo * Hout + ho) * Wout + wo) * SNN_T + t] = acc;
}

// ---------------- CUBA LIF scan, in place (z_t -> spike_t) -----------------
__global__ void snn_cuba_inplace(float* __restrict__ z, int nSeq) {
  const int i = blockIdx.x * blockDim.x + threadIdx.x;
  if (i >= nSeq) return;
  float* p = z + (size_t)i * SNN_T;
  float c = 0.f, v = 0.f;
#pragma unroll 4
  for (int t = 0; t < SNN_T; ++t) {
    c = 0.7f * c + p[t];          // (1 - CURRENT_DECAY)
    v = 0.75f * v + c;            // (1 - VOLTAGE_DECAY)
    const float s = (v >= 1.0f) ? 1.0f : 0.0f;
    p[t] = s;
    v *= (1.0f - s);              // hard reset
  }
}

// CUBA scan fused with write of f16 transposed activations X_T[(b*T+t)*Kp + k].
// z element address = k*strideK + b*strideB + t.  k in [F,Kp) writes zeros.
__global__ void snn_cuba_to_xT(const float* __restrict__ z, _Float16* __restrict__ xT,
                               int F, int Kp, int B, long strideK, long strideB) {
  const int i = blockIdx.x * blockDim.x + threadIdx.x;
  if (i >= B * Kp) return;
  const int k = i % Kp, b = i / Kp;
  _Float16* q = xT + (size_t)b * SNN_T * Kp + k;
  if (k >= F) {
    for (int t = 0; t < SNN_T; ++t) q[(size_t)t * Kp] = (_Float16)0.f;
    return;
  }
  const float* p = z + (size_t)k * strideK + (size_t)b * strideB;
  float c = 0.f, v = 0.f;
  for (int t = 0; t < SNN_T; ++t) {
    c = 0.7f * c + p[t];
    v = 0.75f * v + c;
    const float s = (v >= 1.0f) ? 1.0f : 0.0f;
    q[(size_t)t * Kp] = (_Float16)s;
    v *= (1.0f - s);
  }
}

// Final CUBA: z5 (16 x 1024) -> out (B,11,T) f32
__global__ void snn_cuba_out(const float* __restrict__ z5, float* __restrict__ out) {
  const int i = blockIdx.x * blockDim.x + threadIdx.x;  // b*11 + o
  if (i >= 16 * 11) return;
  const int o = i % 11, b = i / 11;
  const float* p = z5 + (size_t)o * 1024 + (size_t)b * SNN_T;
  float* q = out + (size_t)i * SNN_T;
  float c = 0.f, v = 0.f;
  for (int t = 0; t < SNN_T; ++t) {
    c = 0.7f * c + p[t];
    v = 0.75f * v + c;
    const float s = (v >= 1.0f) ? 1.0f : 0.0f;
    q[t] = s;
    v *= (1.0f - s);
  }
}

// ---------------- WMMA GEMM: D(Mp x N) = A(Mp x K) * Bt^T ------------------
// A row-major f16 (padded, Mp %16==0, K %32==0), Bt is B transposed: (N x K)
// row-major f16 so the B-fragment (one column, two runs of 8 contiguous K) is two 16B
// loads. One wave per 16x16 tile; K-loop of v_wmma_f32_16x16x32_f16.
__global__ void snn_wmma_gemm(const _Float16* __restrict__ A,
                              const _Float16* __restrict__ Bt,
                              float* __restrict__ D, int K, int N) {
  const int lane = threadIdx.x & 31;
  const int half = lane >> 4;        // K-half selector per ISA fragment layout
  const int l16  = lane & 15;
  const int mBase = blockIdx.y * 16;
  const int nBase = blockIdx.x * 16;
  const _Float16* aRow = A  + (size_t)(mBase + l16) * K;
  const _Float16* bRow = Bt + (size_t)(nBase + l16) * K;
  v8f acc = {};
  for (int k0 = 0; k0 < K; k0 += 32) {
    // A 16x32: lane<16 -> K {0..7}+{16..23}; lane>=16 -> K {8..15}+{24..31}
    v8h a0 = *(const v8h*)(aRow + k0 + half * 8);
    v8h a1 = *(const v8h*)(aRow + k0 + 16 + half * 8);
    // B 32x16: lane<16 -> col l16, K {0..7}+{16..23}; lane>=16 -> col l16, K {8..15}+{24..31}
    v8h b0 = *(const v8h*)(bRow + k0 + half * 8);
    v8h b1 = *(const v8h*)(bRow + k0 + 16 + half * 8);
    __builtin_prefetch(aRow + k0 + 32, 0, 1);   // global_prefetch next K tile
    __builtin_prefetch(bRow + k0 + 32, 0, 1);
    v16h av, bv;
#pragma unroll
    for (int j = 0; j < 8; ++j) { av[j] = a0[j]; av[j + 8] = a1[j]; bv[j] = b0[j]; bv[j + 8] = b1[j]; }
    acc = __builtin_amdgcn_wmma_f32_16x16x32_f16(
        false, av, false, bv, (short)0, acc, false, false);
  }
  // D 16x16 f32: lane column = l16, rows = half*8 + v
  float* dOut = D + (size_t)(mBase + half * 8) * N + nBase + l16;
#pragma unroll
  for (int v = 0; v < 8; ++v) dOut[(size_t)v * N] = acc[v];
}

// ---------------------------------------------------------------------------
extern "C" void kernel_launch(void* const* d_in, const int* in_sizes, int n_in,
                              void* d_out, int out_size, void* d_ws, size_t ws_size,
                              hipStream_t stream) {
  (void)in_sizes; (void)n_in; (void)out_size; (void)ws_size;
  const float* x        = (const float*)d_in[0];   // (16,2,128,128,64)
  const float* conv1_v  = (const float*)d_in[1];
  const float* conv1_g  = (const float*)d_in[2];
  const float* conv2_v  = (const float*)d_in[3];
  const float* conv2_g  = (const float*)d_in[4];
  const float* conv3_v  = (const float*)d_in[5];
  const float* conv3_g  = (const float*)d_in[6];
  const float* dense1_v = (const float*)d_in[7];
  const float* dense1_g = (const float*)d_in[8];
  const float* dense2_v = (const float*)d_in[9];
  const float* dense2_g = (const float*)d_in[10];
  float* out = (float*)d_out;                       // (16,11,64)

  // workspace carve (256B aligned)
  char* base = (char*)d_ws;
  size_t off = 0;
  auto carve = [&](size_t bytes) -> void* {
    void* p = base + off;
    off += (bytes + 255) & ~(size_t)255;
    return p;
  };
  float*    w1n = (float*)   carve((size_t)8  * 50  * 4);      // conv1 weights
  float*    w2n = (float*)   carve((size_t)32 * 200 * 4);      // conv2 weights
  float*    w3n = (float*)   carve((size_t)64 * 800 * 4);      // conv3 weights
  _Float16* W1h = (_Float16*)carve((size_t)2064 * 256  * 2);   // dense1, M padded
  _Float16* W2h = (_Float16*)carve((size_t)16   * 2080 * 2);   // dense2, M,K padded
  float*    zs1 = (float*)   carve((size_t)16 * 8  * 32 * 32 * 64 * 4); // z1/s1 in place
  float*    zs2 = (float*)   carve((size_t)16 * 32 * 8  * 8  * 64 * 4); // z2/s2 in place
  float*    z3  = (float*)   carve((size_t)16 * 256 * 64 * 4);
  _Float16* X1T = (_Float16*)carve((size_t)1024 * 256  * 2);   // s3^T f16
  float*    z4  = (float*)   carve((size_t)2064 * 1024 * 4);
  _Float16* X2T = (_Float16*)carve((size_t)1024 * 2080 * 2);   // s4^T f16, K padded
  float*    z5  = (float*)   carve((size_t)16 * 1024 * 4);

  // 1) weight norm
  snn_wnorm_f32<<<8,   64, 0, stream>>>(conv1_v, conv1_g, w1n, 50);
  snn_wnorm_f32<<<32, 128, 0, stream>>>(conv2_v, conv2_g, w2n, 200);
  snn_wnorm_f32<<<64, 256, 0, stream>>>(conv3_v, conv3_g, w3n, 800);
  snn_wnorm_f16<<<129, 256, 0, stream>>>(dense1_v, dense1_g, W1h, 2056, 256, 256);
  snn_wnorm_f16<<<16,  256, 0, stream>>>(dense2_v, dense2_g, W2h, 11, 2056, 2080);

  // 2) conv blocks + CUBA (in place)
  snn_conv<<<dim3(32 * 32, 16 * 8),  64, 0, stream>>>(x,   w1n, zs1, 2, 128, 128, 8, 32, 32);
  snn_cuba_inplace<<<(131072 + 255) / 256, 256, 0, stream>>>(zs1, 131072);
  snn_conv<<<dim3(8 * 8,   16 * 32), 64, 0, stream>>>(zs1, w2n, zs2, 8, 32, 32, 32, 8, 8);
  snn_cuba_inplace<<<(32768 + 255) / 256, 256, 0, stream>>>(zs2, 32768);
  snn_conv<<<dim3(2 * 2,   16 * 64), 64, 0, stream>>>(zs2, w3n, z3,  32, 8, 8, 64, 2, 2);

  // 3) CUBA3 fused with flatten + f16 transpose: z3(b,k,t) -> X1T[(b*64+t)*256+k]
  snn_cuba_to_xT<<<(16 * 256 + 255) / 256, 256, 0, stream>>>(
      z3, X1T, 256, 256, 16, 64L, 256L * 64);

  // 4) dense1 GEMM (2064x1024x256) via v_wmma_f32_16x16x32_f16
  snn_wmma_gemm<<<dim3(1024 / 16, 2064 / 16), 32, 0, stream>>>(W1h, X1T, z4, 256, 1024);

  // 5) CUBA4 fused with f16 transpose + K-pad: z4[o*1024+b*64+t] -> X2T
  snn_cuba_to_xT<<<(16 * 2080 + 255) / 256, 256, 0, stream>>>(
      z4, X2T, 2056, 2080, 16, 1024L, 64L);

  // 6) dense2 GEMM (16x1024x2080) via WMMA
  snn_wmma_gemm<<<dim3(1024 / 16, 1), 32, 0, stream>>>(W2h, X2T, z5, 2080, 1024);

  // 7) final CUBA -> output spikes (16,11,64)
  snn_cuba_out<<<1, 256, 0, stream>>>(z5, out);
}
